// SimpleGCN_56075093016588
// MI455X (gfx1250) — matrix operand order repaired and checked by reading, not compiled, not run on hardware
//
#include <hip/hip_runtime.h>
#include <cstddef>

// ---------------------------------------------------------------------------
// SimpleGCN on MI455X (gfx1250): bf16 WMMA GEMMs. Activations are converted to
// bf16 row-major (contiguous k) so every A fragment is one aligned
// 32B load; W is pre-transposed so every B fragment is one aligned 32B load.
// Ping-pong double-buffered k-loop; float-atomic scatter message passing.
// ---------------------------------------------------------------------------

#define INP 256
#define HID 512
#define OUTF 256
#define BN_EPS 1e-5f

typedef __attribute__((ext_vector_type(16))) __bf16 v16bf;
typedef __attribute__((ext_vector_type(4)))  __bf16 bf16x4;
typedef __attribute__((ext_vector_type(8)))  float  v8f;

__device__ __forceinline__ float gelu_erf(float x) {
    return 0.5f * x * (1.0f + erff(x * 0.70710678118654752f));
}

// -------------------------------- setup kernels ----------------------------

__global__ void fill_kernel(float* __restrict__ p, float v, int n) {
    int i = blockIdx.x * blockDim.x + threadIdx.x;
    if (i < n) p[i] = v;
}

__global__ void deg_kernel(float* __restrict__ deg, const int* __restrict__ col, int E) {
    int e = blockIdx.x * blockDim.x + threadIdx.x;
    if (e < E) atomicAdd(&deg[col[e]], 1.0f);
}

__global__ void rsqrt_kernel(float* __restrict__ d, int n) {
    int i = blockIdx.x * blockDim.x + threadIdx.x;
    if (i < n) {
        float v = d[i];
        d[i] = (v > 0.0f) ? rsqrtf(v) : 0.0f;
    }
}

__global__ void norm_kernel(float* __restrict__ nrm, const float* __restrict__ dinv,
                            const int* __restrict__ row, const int* __restrict__ col, int E) {
    int e = blockIdx.x * blockDim.x + threadIdx.x;
    if (e < E) nrm[e] = dinv[row[e]] * dinv[col[e]];
}

// ------------------------- operand preparation -----------------------------

// Hb = bf16(gelu?(src)) in plain row-major order (group map is identity):
// within each block of 32, 8-element groups keep order [g0 g1 g2 g3] so that
// a WMMA A-fragment lane-half reads one contiguous 32B chunk.
// kshift = log2(K); processes float4 (4 consecutive k, always within a group).
__global__ void cvt_bf16_kernel(const float* __restrict__ src, __bf16* __restrict__ dst,
                                int n4, int kshift, int do_gelu) {
    int i = blockIdx.x * blockDim.x + threadIdx.x;
    if (i >= n4) return;
    float4 v = ((const float4*)src)[i];
    if (do_gelu) {
        v.x = gelu_erf(v.x); v.y = gelu_erf(v.y);
        v.z = gelu_erf(v.z); v.w = gelu_erf(v.w);
    }
    bf16x4 o;
    o[0] = (__bf16)v.x; o[1] = (__bf16)v.y; o[2] = (__bf16)v.z; o[3] = (__bf16)v.w;

    const int el   = i << 2;                        // linear element index
    const int kmask = (1 << kshift) - 1;
    const int r    = el >> kshift;
    const int k    = el & kmask;
    const int grp  = (k >> 3) & 3;
    const int ngrp = grp;                           // identity: contiguous k
    const int dk   = (k & ~31) | (ngrp << 3) | (k & 7);
    const int didx = ((r << kshift) | dk) >> 2;     // bf16x4 granularity
    ((bf16x4*)dst)[didx] = o;
}

// Wt[n*K + k] = bf16( W[k*Nc + n] )   (transpose to [Nc,K] for contiguous B frags)
__global__ void wt_bf16_kernel(const float* __restrict__ W, __bf16* __restrict__ Wt,
                               int K, int Nc) {
    int idx = blockIdx.x * blockDim.x + threadIdx.x;
    if (idx >= K * Nc) return;
    int n = idx / K;
    int k = idx - n * K;
    Wt[idx] = (__bf16)W[(size_t)k * Nc + n];
}

// -------------------------------- WMMA GEMM --------------------------------
// C[M,Nc] = A[M,K] @ B[K,Nc] (+bias). A bf16 row-major, contiguous k (see
// cvt_bf16_kernel), Bt = B^T bf16 [Nc,K]. Block: 256 threads (8 waves)
// covering 16*wavesM rows x (8/wavesM)*64 cols (block spans the full Nc).
// Wave tile: 16 x 64 (4 accumulators, one shared A fragment). Every fragment
// is a single aligned 32B (v16bf) load; two fragment sets ping-pong by k-step
// parity (no register copies). Requires K % 64 == 0, K >= 128 (K in {256,512}).
__global__ __launch_bounds__(256)
void gemm_bf16t_wmma(const __bf16* __restrict__ A, const __bf16* __restrict__ Bt,
                     float* __restrict__ C, const float* __restrict__ bias,
                     int M, int K, int Nc, int wavesM) {
    const int wave = threadIdx.x >> 5;
    const int lane = threadIdx.x & 31;
    const int half = lane >> 4;      // 0: lanes 0-15, 1: lanes 16-31
    const int l16  = lane & 15;
    const int wavesN = 8 / wavesM;                     // waves along N
    const int row0 = blockIdx.x * 16 * wavesM + (wave / wavesN) * 16;
    const int c0   = (wave % wavesN) * 64;             // Nc == wavesN*64

    int arow = row0 + l16;
    if (arow >= M) arow = M - 1;                       // clamp loads; stores guarded
    const __bf16* Ar = A + (size_t)arow * K + (half << 4);
    const __bf16* Br = Bt + (size_t)(c0 + l16) * K + (half << 4);

    // A fragment 16x32 (row-major, half h holds K=k0+16h..+15): one contiguous 32B chunk per lane.
    auto load_a = [&](int k0) -> v16bf {
        return *(const v16bf*)(Ar + k0);
    };
    // B fragment 32x16 (tile t): lane = column n, half h holds K=k0+16h..+15,
    // contiguous in Bt[n,:]: one 32B chunk per lane.
    auto load_b = [&](int k0, int t) -> v16bf {
        return *(const v16bf*)(Br + (size_t)(t * 16) * K + k0);
    };

    v8f acc[4] = {};

    // prologue: fill both fragment sets (k-steps 0 and 32)
    v16bf aA = load_a(0);
    v16bf bA[4];
#pragma unroll
    for (int t = 0; t < 4; ++t) bA[t] = load_b(0, t);
    v16bf aB = load_a(32);
    v16bf bB[4];
#pragma unroll
    for (int t = 0; t < 4; ++t) bB[t] = load_b(32, t);

    // steady state: consume set, immediately refill it for k+64 (no reg copies)
    int k0 = 0;
    for (; k0 + 64 < K; k0 += 64) {
#pragma unroll
        for (int t = 0; t < 4; ++t)
            acc[t] = __builtin_amdgcn_wmma_f32_16x16x32_bf16(false, aA, false, bA[t],
                                                             (short)0, acc[t], false, false);
        aA = load_a(k0 + 64);
#pragma unroll
        for (int t = 0; t < 4; ++t) bA[t] = load_b(k0 + 64, t);
#pragma unroll
        for (int t = 0; t < 4; ++t)
            acc[t] = __builtin_amdgcn_wmma_f32_16x16x32_bf16(false, aB, false, bB[t],
                                                             (short)0, acc[t], false, false);
        aB = load_a(k0 + 96);
#pragma unroll
        for (int t = 0; t < 4; ++t) bB[t] = load_b(k0 + 96, t);
    }
    // epilogue: last two k-steps
#pragma unroll
    for (int t = 0; t < 4; ++t)
        acc[t] = __builtin_amdgcn_wmma_f32_16x16x32_bf16(false, aA, false, bA[t],
                                                         (short)0, acc[t], false, false);
#pragma unroll
    for (int t = 0; t < 4; ++t)
        acc[t] = __builtin_amdgcn_wmma_f32_16x16x32_bf16(false, aB, false, bB[t],
                                                         (short)0, acc[t], false, false);

    // C/D layout: lane n = l16, VGPR i -> M = i + 8*half
    const int crow = row0 + (half << 3);
#pragma unroll
    for (int t = 0; t < 4; ++t) {
        const int cc = c0 + t * 16 + l16;
        const float bv = bias ? bias[cc] : 0.0f;
#pragma unroll
        for (int i = 0; i < 8; ++i) {
            if (crow + i < M)
                C[(size_t)(crow + i) * Nc + cc] = acc[t][i] + bv;
        }
    }
}

// ------------------------- message passing kernels -------------------------

// H[i,f] = b[f] + HW[i,f] * dinv[i]^2   (self-loop term + bias, float4)
__global__ void self_bias_kernel(float* __restrict__ H, const float* __restrict__ HW,
                                 const float* __restrict__ dinv, const float* __restrict__ b,
                                 int Nn) {
    int idx = blockIdx.x * blockDim.x + threadIdx.x;   // Nn * (HID/4)
    if (idx >= Nn * (HID / 4)) return;
    int i = idx >> 7;
    int c = (idx & 127) << 2;
    float s = dinv[i]; s = s * s;
    const float4 v = *(const float4*)(HW + (size_t)i * HID + c);
    const float4 bb = *(const float4*)(b + c);
    float4 o;
    o.x = bb.x + v.x * s; o.y = bb.y + v.y * s;
    o.z = bb.z + v.z * s; o.w = bb.w + v.w * s;
    *(float4*)(H + (size_t)i * HID + c) = o;
}

// H[col[e], :] += HW[row[e], :] * nrm[e]   (float4 chunks, f32 atomics)
__global__ void scatter_kernel(float* __restrict__ H, const float* __restrict__ HW,
                               const int* __restrict__ row, const int* __restrict__ col,
                               const float* __restrict__ nrm, int E) {
    int idx = blockIdx.x * blockDim.x + threadIdx.x;   // E * (HID/4)
    if (idx >= E * (HID / 4)) return;
    int e = idx >> 7;
    int c = (idx & 127) << 2;
    const int r = row[e], cl = col[e];
    const float w = nrm[e];
    const float4 v = *(const float4*)(HW + (size_t)r * HID + c);
    float* dst = H + (size_t)cl * HID + c;
    atomicAdd(dst + 0, v.x * w);
    atomicAdd(dst + 1, v.y * w);
    atomicAdd(dst + 2, v.z * w);
    atomicAdd(dst + 3, v.w * w);
}

// ------------------------------- batchnorm ---------------------------------

#define BN_ROWS 256
__global__ __launch_bounds__(512)
void bn_stats_kernel(const float* __restrict__ H, float* __restrict__ stats, int M) {
    int f = threadIdx.x;                 // 512 threads == 512 features
    int r0 = blockIdx.x * BN_ROWS;
    int rend = r0 + BN_ROWS; if (rend > M) rend = M;
    float s = 0.0f, q = 0.0f;
    for (int r = r0; r < rend; ++r) {
        float v = H[(size_t)r * HID + f];
        s += v; q += v * v;
    }
    atomicAdd(&stats[f], s);
    atomicAdd(&stats[HID + f], q);
}

__global__ void bn_apply_gelu_kernel(float* __restrict__ H, const float* __restrict__ stats,
                                     const float* __restrict__ g, const float* __restrict__ bta,
                                     int Nn) {
    int idx = blockIdx.x * blockDim.x + threadIdx.x;   // Nn * (HID/4)
    if (idx >= Nn * (HID / 4)) return;
    int i = idx >> 7;
    int c = (idx & 127) << 2;
    const float invM = 1.0f / (float)Nn;
    float4 v = *(const float4*)(H + (size_t)i * HID + c);
    float o[4] = {v.x, v.y, v.z, v.w};
#pragma unroll
    for (int j = 0; j < 4; ++j) {
        int f = c + j;
        float mu  = stats[f] * invM;
        float var = stats[HID + f] * invM - mu * mu;
        float y = (o[j] - mu) * rsqrtf(var + BN_EPS) * g[f] + bta[f];
        o[j] = gelu_erf(y);
    }
    v.x = o[0]; v.y = o[1]; v.z = o[2]; v.w = o[3];
    *(float4*)(H + (size_t)i * HID + c) = v;
}

// ------------------------------- launcher ----------------------------------

extern "C" void kernel_launch(void* const* d_in, const int* in_sizes, int n_in,
                              void* d_out, int out_size, void* d_ws, size_t ws_size,
                              hipStream_t stream) {
    const float* x       = (const float*)d_in[0];
    const int*   ei      = (const int*)d_in[1];
    const float* W_first = (const float*)d_in[2];
    const float* b_first = (const float*)d_in[3];
    const float* W_rest  = (const float*)d_in[4];
    const float* b_rest  = (const float*)d_in[5];
    const float* bn_g    = (const float*)d_in[6];
    const float* bn_b    = (const float*)d_in[7];
    const float* out_W   = (const float*)d_in[8];
    const float* out_b   = (const float*)d_in[9];

    const int Nn = in_sizes[0] / INP;      // 50000 (multiple of 16)
    const int E  = in_sizes[1] / 2;        // 1.6M
    const int* erow = ei;
    const int* ecol = ei + E;

    // workspace layout
    float*  H     = (float*)d_ws;                       // Nn*HID f32
    float*  HW    = H + (size_t)Nn * HID;               // Nn*HID f32
    float*  nrm   = HW + (size_t)Nn * HID;              // E f32
    float*  dinv  = nrm + E;                            // Nn f32
    float*  stats = dinv + Nn;                          // 2*HID f32
    __bf16* Hb    = (__bf16*)(stats + 2 * HID);         // Nn*HID bf16
    __bf16* Wtb   = Hb + (size_t)Nn * HID;              // HID*HID bf16 (max)
    (void)ws_size; (void)n_in; (void)out_size;

    const int TB = 256;
    const int nElem4 = Nn * (HID / 4);
    const int eElem4 = E * (HID / 4);

    // ---- degrees / symmetric normalization ----
    fill_kernel<<<(Nn + TB - 1) / TB, TB, 0, stream>>>(dinv, 1.0f, Nn);  // self-loop
    deg_kernel<<<(E + TB - 1) / TB, TB, 0, stream>>>(dinv, ecol, E);
    rsqrt_kernel<<<(Nn + TB - 1) / TB, TB, 0, stream>>>(dinv, Nn);
    norm_kernel<<<(E + TB - 1) / TB, TB, 0, stream>>>(nrm, dinv, erow, ecol, E);

    // one conv: Hb = bf16(gelu?(src)); Wtb = W^T; HW = Hb@W;
    //           H = b + HW*dinv^2; scatter edges into H
    auto do_conv = [&](const float* src, int K, const float* W, const float* b, int do_gelu) {
        const int kshift = (K == 512) ? 9 : 8;
        cvt_bf16_kernel<<<(Nn * K / 4 + TB - 1) / TB, TB, 0, stream>>>(
            src, Hb, Nn * K / 4, kshift, do_gelu);
        wt_bf16_kernel<<<(K * HID + TB - 1) / TB, TB, 0, stream>>>(W, Wtb, K, HID);
        gemm_bf16t_wmma<<<dim3((Nn + 15) / 16), TB, 0, stream>>>(Hb, Wtb, HW, nullptr,
                                                                 Nn, K, HID, 1);
        self_bias_kernel<<<(nElem4 + TB - 1) / TB, TB, 0, stream>>>(H, HW, dinv, b, Nn);
        scatter_kernel<<<(eElem4 + TB - 1) / TB, TB, 0, stream>>>(H, HW, erow, ecol, nrm, E);
    };
    auto Wr = [&](int i) { return W_rest + (size_t)i * HID * HID; };
    auto br = [&](int i) { return b_rest + (size_t)i * HID; };

    // L0..L2
    do_conv(x, INP, W_first, b_first, 0);
    do_conv(H, HID, Wr(0), br(0), 1);
    do_conv(H, HID, Wr(1), br(1), 1);

    int wi = 2;
    for (int blk = 0; blk < 3; ++blk) {
        do_conv(H, HID, Wr(wi), br(wi), 0); ++wi;      // no activation before
        do_conv(H, HID, Wr(wi), br(wi), 1); ++wi;
        do_conv(H, HID, Wr(wi), br(wi), 1); ++wi;
        if (blk != 2) {
            fill_kernel<<<(2 * HID + TB - 1) / TB, TB, 0, stream>>>(stats, 0.0f, 2 * HID);
            bn_stats_kernel<<<(Nn + BN_ROWS - 1) / BN_ROWS, 512, 0, stream>>>(H, stats, Nn);
            bn_apply_gelu_kernel<<<(nElem4 + TB - 1) / TB, TB, 0, stream>>>(
                H, stats, bn_g + blk * HID, bn_b + blk * HID, Nn);
        }
    }

    // output projection: Hb = bf16(H); Wtb = out_W^T; d_out = Hb@out_W + out_b
    cvt_bf16_kernel<<<(nElem4 + TB - 1) / TB, TB, 0, stream>>>(H, Hb, nElem4, 9, 0);
    wt_bf16_kernel<<<(HID * OUTF + TB - 1) / TB, TB, 0, stream>>>(out_W, Wtb, HID, OUTF);
    gemm_bf16t_wmma<<<dim3((Nn + 31) / 32), TB, 0, stream>>>(Hb, Wtb, (float*)d_out, out_b,
                                                             Nn, HID, OUTF, 2);
}
